// TransformerBlock_37538014167312
// MI455X (gfx1250) — matrix operand order repaired and checked by reading, not compiled, not run on hardware
//
#include <hip/hip_runtime.h>
#include <hip/hip_bf16.h>
#include <math.h>

// ---------------- problem constants (match reference) ----------------
constexpr int BATCH = 4;
constexpr int SEQ   = 2048;
constexpr int CDIM  = 768;
constexpr int NHEAD = 12;
constexpr int HDIM  = 64;          // CDIM / NHEAD
constexpr int WLOC  = 7;
constexpr int HID   = 3072;
constexpr int HID2  = 1536;
constexpr int MROWS = BATCH * SEQ; // 8192

// ---------------- CDNA5 vector types ----------------
typedef __attribute__((ext_vector_type(16))) __bf16 v16bf;
typedef __attribute__((ext_vector_type(8)))  float  v8f;
typedef __attribute__((ext_vector_type(4)))  int    v4i;

#define AS1 __attribute__((address_space(1)))
#define AS3 __attribute__((address_space(3)))

// ---- CDNA5 async global->LDS path (GLOBAL_LOAD_ASYNC_TO_LDS_B128, ASYNCcnt) ----
#if defined(__has_builtin)
#if __has_builtin(__builtin_amdgcn_global_load_async_to_lds_b128)
#define USE_ASYNC_LDS 1
#endif
#endif
#ifndef USE_ASYNC_LDS
#define USE_ASYNC_LDS 0
#endif

#if USE_ASYNC_LDS
#if __has_builtin(__builtin_amdgcn_s_wait_asynccnt)
#define WAIT_ASYNC0() __builtin_amdgcn_s_wait_asynccnt(0)
#else
#define WAIT_ASYNC0() asm volatile("s_wait_asynccnt 0" ::: "memory")
#endif
#endif

__device__ __forceinline__ unsigned short f2bf(float f) {
    unsigned int u = __float_as_uint(f);
    u += 0x7fffu + ((u >> 16) & 1u);   // round-to-nearest-even
    return (unsigned short)(u >> 16);
}

// ---------------- weight cast+transpose fp32 [K,N] -> bf16 [N,K] ----------------
__global__ void cast_transpose_bf16_kernel(const float* __restrict__ src,
                                           unsigned short* __restrict__ dst,
                                           int K, int N) {
    int idx = blockIdx.x * blockDim.x + threadIdx.x;  // coalesced over src
    if (idx >= K * N) return;
    int k = idx / N;
    int n = idx - k * N;
    dst[(size_t)n * K + k] = f2bf(src[idx]);
}

// ---------------- layernorm (fp32 in) -> bf16 out ----------------
template <int COLS>
__global__ __launch_bounds__(256) void layernorm_bf16_kernel(
    const float* __restrict__ x, const float* __restrict__ w,
    const float* __restrict__ b, unsigned short* __restrict__ out) {
    __shared__ float red[256];
    const int row = blockIdx.x;
    const float* xr = x + (size_t)row * COLS;

    float s = 0.f;
    for (int c = threadIdx.x; c < COLS; c += 256) s += xr[c];
    red[threadIdx.x] = s;
    __syncthreads();
    for (int st = 128; st > 0; st >>= 1) {
        if (threadIdx.x < st) red[threadIdx.x] += red[threadIdx.x + st];
        __syncthreads();
    }
    const float mu = red[0] / COLS;
    __syncthreads();

    float v = 0.f;
    for (int c = threadIdx.x; c < COLS; c += 256) {
        float d = xr[c] - mu;
        v += d * d;
    }
    red[threadIdx.x] = v;
    __syncthreads();
    for (int st = 128; st > 0; st >>= 1) {
        if (threadIdx.x < st) red[threadIdx.x] += red[threadIdx.x + st];
        __syncthreads();
    }
    const float rstd = rsqrtf(red[0] / COLS + 1e-5f);

    unsigned short* orow = out + (size_t)row * COLS;
    for (int c = threadIdx.x; c < COLS; c += 256)
        orow[c] = f2bf((xr[c] - mu) * rstd * w[c] + b[c]);
}

// ---------------- bf16 WMMA GEMM: out = epilogue(A[M,K] @ W) with W given as Wt[N,K] ----
// 256 threads = 8 wave32. Block tile 128x128, wave tile 32x64, K step 32.
// Double-buffered LDS; tiles staged with async global->LDS copies when available.
constexpr int TM = 128, TN = 128, TK = 32;
constexpr int LDS_STRIDE = 40;  // 32 + 8 pad, keeps 16B alignment, spreads banks

template <bool HAS_BIAS, bool DO_GELU, bool HAS_RES, bool OUT_BF16>
__global__ __launch_bounds__(256) void gemm_wmma_bf16(
    const unsigned short* __restrict__ A,   // [M,K]  bf16 row-major
    const unsigned short* __restrict__ Bt,  // [N,K]  bf16 row-major (pre-transposed)
    const float* __restrict__ bias,         // [N]
    const float* __restrict__ resid,        // [M,N]  fp32
    void* __restrict__ outp,                // [M,N]  fp32 or bf16
    int M, int Nc, int K) {
    __shared__ __align__(16) unsigned short As[2][TM * LDS_STRIDE];
    __shared__ __align__(16) unsigned short Bs[2][TN * LDS_STRIDE];

    const int tid  = threadIdx.x;
    const int wave = tid >> 5;
    const int lane = tid & 31;
    const int hi   = lane >> 4;        // K-half selector (contiguous k order used here)
    const int l16  = lane & 15;
    const int wm   = (wave >> 1) * 32; // wave M offset: 0,32,64,96
    const int wn   = (wave & 1) * 64;  // wave N offset: 0,64
    const int m0   = blockIdx.y * TM;
    const int n0   = blockIdx.x * TN;

    const v8f zero = {};
    v8f acc[2][4];
#pragma unroll
    for (int mi = 0; mi < 2; mi++)
#pragma unroll
        for (int ni = 0; ni < 4; ni++) acc[mi][ni] = zero;

    // staging geometry: 256 threads cover a 128x32 bf16 tile; 32B (2x b128) per thread
    const int srow  = tid >> 1;
    const int shalf = (tid & 1) * 16;
    unsigned short* gA = const_cast<unsigned short*>(A)  + (size_t)(m0 + srow) * K + shalf;
    unsigned short* gB = const_cast<unsigned short*>(Bt) + (size_t)(n0 + srow) * K + shalf;
    const int nk = K / TK;

#if USE_ASYNC_LDS
    auto stage_async = [&](int kt, int buf) {
        unsigned short* ga = gA + kt * TK;
        unsigned short* gb = gB + kt * TK;
        unsigned short* la = &As[buf][srow * LDS_STRIDE + shalf];
        unsigned short* lb = &Bs[buf][srow * LDS_STRIDE + shalf];
        __builtin_amdgcn_global_load_async_to_lds_b128((AS1 v4i*)ga,       (AS3 v4i*)la,       0, 0);
        __builtin_amdgcn_global_load_async_to_lds_b128((AS1 v4i*)(ga + 8), ((AS3 v4i*)la) + 1, 0, 0);
        __builtin_amdgcn_global_load_async_to_lds_b128((AS1 v4i*)gb,       (AS3 v4i*)lb,       0, 0);
        __builtin_amdgcn_global_load_async_to_lds_b128((AS1 v4i*)(gb + 8), ((AS3 v4i*)lb) + 1, 0, 0);
    };
    // prologue: stage tile 0 into buffer 0
    stage_async(0, 0);
    WAIT_ASYNC0();
#else
    {   // prologue: synchronous stage of tile 0
        const v4i* pga = reinterpret_cast<const v4i*>(gA);
        const v4i* pgb = reinterpret_cast<const v4i*>(gB);
        v4i a0 = pga[0], a1 = pga[1], b0 = pgb[0], b1 = pgb[1];
        v4i* la = reinterpret_cast<v4i*>(&As[0][srow * LDS_STRIDE + shalf]);
        v4i* lb = reinterpret_cast<v4i*>(&Bs[0][srow * LDS_STRIDE + shalf]);
        la[0] = a0; la[1] = a1; lb[0] = b0; lb[1] = b1;
    }
#endif
    __syncthreads();

    for (int kt = 0; kt < nk; ++kt) {
        const int cur = kt & 1;
        const int nxt = cur ^ 1;
        const bool more = (kt + 1) < nk;

#if USE_ASYNC_LDS
        if (more) stage_async(kt + 1, nxt);   // async copies run under the WMMAs
#else
        v4i ra0 = {}, ra1 = {}, rb0 = {}, rb1 = {};
        if (more) {                            // issue global loads early (in flight during compute)
            const v4i* pga = reinterpret_cast<const v4i*>(gA + (kt + 1) * TK);
            const v4i* pgb = reinterpret_cast<const v4i*>(gB + (kt + 1) * TK);
            ra0 = pga[0]; ra1 = pga[1]; rb0 = pgb[0]; rb1 = pgb[1];
        }
#endif

        // ---- fragments in contiguous k order (K = 16*hi + i), from current LDS buffer ----
        union Frag { v16bf v; v4i q[2]; } afr[2], bfr[4];
#pragma unroll
        for (int mi = 0; mi < 2; mi++) {
            // A 16x32: lane(l16)=M row; elem i -> K = 16*hi + i
            const unsigned short* base = &As[cur][(wm + mi * 16 + l16) * LDS_STRIDE];
            afr[mi].q[0] = *reinterpret_cast<const v4i*>(base + 16 * hi);
            afr[mi].q[1] = *reinterpret_cast<const v4i*>(base + 16 * hi + 8);
        }
#pragma unroll
        for (int ni = 0; ni < 4; ni++) {
            // B 32x16: lane(l16)=N col; elem i -> K = 16*hi + i (contiguous in Bs[n][k])
            const unsigned short* base = &Bs[cur][(wn + ni * 16 + l16) * LDS_STRIDE + 16 * hi];
            bfr[ni].q[0] = *reinterpret_cast<const v4i*>(base);
            bfr[ni].q[1] = *reinterpret_cast<const v4i*>(base + 8);
        }
#pragma unroll
        for (int mi = 0; mi < 2; mi++)
#pragma unroll
            for (int ni = 0; ni < 4; ni++)
                acc[mi][ni] = __builtin_amdgcn_wmma_f32_16x16x32_bf16(
                    false, afr[mi].v, false, bfr[ni].v,
                    (short)0, acc[mi][ni], false, false);

#if USE_ASYNC_LDS
        if (more) WAIT_ASYNC0();              // LDS writes of next tile landed
#else
        if (more) {
            v4i* la = reinterpret_cast<v4i*>(&As[nxt][srow * LDS_STRIDE + shalf]);
            v4i* lb = reinterpret_cast<v4i*>(&Bs[nxt][srow * LDS_STRIDE + shalf]);
            la[0] = ra0; la[1] = ra1; lb[0] = rb0; lb[1] = rb1;
        }
#endif
        __syncthreads();
    }

    // ---- epilogue: C/D layout -> row = base + r + 8*hi, col = base + l16 ----
#pragma unroll
    for (int mi = 0; mi < 2; mi++) {
#pragma unroll
        for (int ni = 0; ni < 4; ni++) {
#pragma unroll
            for (int r = 0; r < 8; r++) {
                const int row = m0 + wm + mi * 16 + hi * 8 + r;
                const int col = n0 + wn + ni * 16 + l16;
                float v = acc[mi][ni][r];
                if (HAS_BIAS) v += bias[col];
                if (DO_GELU)  v = 0.5f * v * (1.0f + erff(v * 0.70710678118654752f));
                if (HAS_RES)  v += resid[(size_t)row * Nc + col];
                if (OUT_BF16)
                    ((unsigned short*)outp)[(size_t)row * Nc + col] = f2bf(v);
                else
                    ((float*)outp)[(size_t)row * Nc + col] = v;
            }
        }
    }
}

// ---------------- local-window attention (|i-j| <= 7), VALU ----------------
// qkv: fp32 [B*N, 3*C], col layout [part(3)][head(12)][d(64)].
// out: bf16 [B*N, C].  One thread per (b, h, n) row.
__global__ __launch_bounds__(256) void attention_local_kernel(
    const float* __restrict__ qkv, unsigned short* __restrict__ out) {
    const int gid = blockIdx.x * blockDim.x + threadIdx.x;
    const int n  = gid % SEQ;
    const int bh = gid / SEQ;
    const int h  = bh % NHEAD;
    const int b  = bh / NHEAD;

    const float* q = qkv + (size_t)(b * SEQ + n) * (3 * CDIM) + h * HDIM;
    float qr[HDIM];
#pragma unroll
    for (int d = 0; d < HDIM; d++) qr[d] = q[d];

    constexpr int WIN = 2 * WLOC + 1;
    const float scale = 0.125f;  // 1/sqrt(64)
    float sc[WIN];
    float mx = -3.0e30f;
#pragma unroll
    for (int jj = 0; jj < WIN; jj++) {
        const int j   = n - WLOC + jj;
        const bool ok = (j >= 0) && (j < SEQ);
        const int jc  = ok ? j : n;
        const float* k = qkv + (size_t)(b * SEQ + jc) * (3 * CDIM) + CDIM + h * HDIM;
        float dot = 0.f;
#pragma unroll
        for (int d = 0; d < HDIM; d++) dot += qr[d] * k[d];
        sc[jj] = ok ? dot * scale : -3.0e30f;
        mx = fmaxf(mx, sc[jj]);
    }
    float ssum = 0.f;
#pragma unroll
    for (int jj = 0; jj < WIN; jj++) {
        sc[jj] = __expf(sc[jj] - mx);
        ssum += sc[jj];
    }
    const float inv = 1.0f / ssum;

    float o[HDIM];
#pragma unroll
    for (int d = 0; d < HDIM; d++) o[d] = 0.f;
#pragma unroll
    for (int jj = 0; jj < WIN; jj++) {
        const int j   = n - WLOC + jj;
        const bool ok = (j >= 0) && (j < SEQ);
        const int jc  = ok ? j : n;
        const float* v = qkv + (size_t)(b * SEQ + jc) * (3 * CDIM) + 2 * CDIM + h * HDIM;
        const float w = ok ? sc[jj] * inv : 0.f;
#pragma unroll
        for (int d = 0; d < HDIM; d++) o[d] += w * v[d];
    }

    unsigned short* orow = out + (size_t)(b * SEQ + n) * CDIM + h * HDIM;
#pragma unroll
    for (int d = 0; d < HDIM; d++) orow[d] = f2bf(o[d]);
}

// ---------------- launch ----------------
extern "C" void kernel_launch(void* const* d_in, const int* in_sizes, int n_in,
                              void* d_out, int out_size, void* d_ws, size_t ws_size,
                              hipStream_t stream) {
    (void)in_sizes; (void)n_in; (void)out_size; (void)ws_size;

    const float* x      = (const float*)d_in[0];
    const float* ln1_w  = (const float*)d_in[1];
    const float* ln1_b  = (const float*)d_in[2];
    const float* qkv_w  = (const float*)d_in[3];
    const float* proj_w = (const float*)d_in[4];
    const float* proj_b = (const float*)d_in[5];
    const float* ln2_w  = (const float*)d_in[6];
    const float* ln2_b  = (const float*)d_in[7];
    const float* fc1_w  = (const float*)d_in[8];
    const float* fc1_b  = (const float*)d_in[9];
    const float* fc2_w  = (const float*)d_in[10];
    const float* fc2_b  = (const float*)d_in[11];
    const float* fc3_w  = (const float*)d_in[12];
    const float* fc3_b  = (const float*)d_in[13];

    char* ws = (char*)d_ws;
    size_t off = 0;
    auto alloc = [&](size_t bytes) -> char* {
        char* p = ws + off;
        off += (bytes + 255) & ~(size_t)255;
        return p;
    };
    unsigned short* xn1  = (unsigned short*)alloc((size_t)MROWS * CDIM * 2);
    float*          qkvb = (float*)         alloc((size_t)MROWS * 3 * CDIM * 4);
    unsigned short* attn = (unsigned short*)alloc((size_t)MROWS * CDIM * 2);
    float*          xres = (float*)         alloc((size_t)MROWS * CDIM * 4);
    unsigned short* xn2  = (unsigned short*)alloc((size_t)MROWS * CDIM * 2);
    unsigned short* h1   = (unsigned short*)alloc((size_t)MROWS * HID * 2);
    unsigned short* h2   = (unsigned short*)alloc((size_t)MROWS * HID2 * 2);
    unsigned short* wqb  = (unsigned short*)alloc((size_t)CDIM * 3 * CDIM * 2);
    unsigned short* wpb  = (unsigned short*)alloc((size_t)CDIM * CDIM * 2);
    unsigned short* w1b  = (unsigned short*)alloc((size_t)CDIM * HID * 2);
    unsigned short* w2b  = (unsigned short*)alloc((size_t)HID * HID2 * 2);
    unsigned short* w3b  = (unsigned short*)alloc((size_t)HID2 * CDIM * 2);

    // weights -> bf16, transposed to [N,K] so GEMM B-tiles are contiguous
    auto castT = [&](const float* s, unsigned short* d, int K, int N) {
        cast_transpose_bf16_kernel<<<(K * N + 255) / 256, 256, 0, stream>>>(s, d, K, N);
    };
    castT(qkv_w,  wqb, CDIM, 3 * CDIM);
    castT(proj_w, wpb, CDIM, CDIM);
    castT(fc1_w,  w1b, CDIM, HID);
    castT(fc2_w,  w2b, HID,  HID2);
    castT(fc3_w,  w3b, HID2, CDIM);

    // attention branch
    layernorm_bf16_kernel<CDIM><<<MROWS, 256, 0, stream>>>(x, ln1_w, ln1_b, xn1);
    gemm_wmma_bf16<false, false, false, false>
        <<<dim3(3 * CDIM / TN, MROWS / TM), 256, 0, stream>>>(
            xn1, wqb, nullptr, nullptr, qkvb, MROWS, 3 * CDIM, CDIM);
    attention_local_kernel<<<(BATCH * NHEAD * SEQ) / 256, 256, 0, stream>>>(qkvb, attn);
    gemm_wmma_bf16<true, false, true, false>
        <<<dim3(CDIM / TN, MROWS / TM), 256, 0, stream>>>(
            attn, wpb, proj_b, x, xres, MROWS, CDIM, CDIM);

    // MLP branch
    layernorm_bf16_kernel<CDIM><<<MROWS, 256, 0, stream>>>(xres, ln2_w, ln2_b, xn2);
    gemm_wmma_bf16<true, true, false, true>
        <<<dim3(HID / TN, MROWS / TM), 256, 0, stream>>>(
            xn2, w1b, fc1_b, nullptr, h1, MROWS, HID, CDIM);
    gemm_wmma_bf16<true, true, false, true>
        <<<dim3(HID2 / TN, MROWS / TM), 256, 0, stream>>>(
            h1, w2b, fc2_b, nullptr, h2, MROWS, HID2, HID);
    gemm_wmma_bf16<true, false, true, false>
        <<<dim3(CDIM / TN, MROWS / TM), 256, 0, stream>>>(
            h2, w3b, fc3_b, xres, (float*)d_out, MROWS, CDIM, HID2);
}
